// WMSA_11888469475357
// MI455X (gfx1250) — matrix operand order repaired and checked by reading, not compiled, not run on hardware
//
#include <hip/hip_runtime.h>
#include <hip/hip_bf16.h>

typedef __bf16 bf16;
typedef __attribute__((ext_vector_type(16))) __bf16 v16bf;
typedef __attribute__((ext_vector_type(8)))  float  v8f;

#define WIN_N 49
#define C_DIM 192
#define NH    6
#define HD    32
#define NPAD  64
#define SXP   200   // bf16 row stride for X/Q/K/O and weight panels (100 dwords: conflict-free)
#define SVT   72    // bf16 row stride for V^T (36 dwords: conflict-free)
#define SSIM  68    // f32 row stride for sim (68 dwords: conflict-free)

union FragBits {
    v16bf v;
    uint4 q[2];
};

// A fragment (16x32 bf16, M x K), source row-major [row][k]:
// lanes 0-15: row=row0+lane, K = k0+0..7 and k0+16..23
// lanes 16-31: row=row0+lane-16, K = k0+8..15 and k0+24..31
__device__ __forceinline__ v16bf load_a_lds(const bf16* base, int row0, int stride, int k0, int lane) {
    int l = lane & 15, hi = lane >> 4;
    const bf16* p = base + (row0 + l) * stride + k0 + hi * 8;
    FragBits f;
    f.q[0] = *reinterpret_cast<const uint4*>(p);
    f.q[1] = *reinterpret_cast<const uint4*>(p + 16);
    return f.v;
}

// B fragment (32x16 bf16, K x N), source stored as [n][k] row-major (i.e. B^T):
// lanes 0-15: col=col0+lane, K = k0+0..7 and k0+16..23 ; lanes 16-31: col=col0+lane-16, K = k0+8..15 and k0+24..31
__device__ __forceinline__ v16bf load_b_lds(const bf16* base, int col0, int stride, int k0, int lane) {
    int l = lane & 15, hi = lane >> 4;
    const bf16* p = base + (col0 + l) * stride + k0 + hi * 8;
    FragBits f;
    f.q[0] = *reinterpret_cast<const uint4*>(p);
    f.q[1] = *reinterpret_cast<const uint4*>(p + 16);
    return f.v;
}

// A fragment from f32 LDS (softmax probabilities), converted to bf16 on the fly
__device__ __forceinline__ v16bf load_a_lds_f32(const float* base, int row0, int stride, int k0, int lane) {
    int l = lane & 15, hi = lane >> 4;
    const float* p = base + (row0 + l) * stride + k0 + hi * 8;
    v16bf r;
#pragma unroll
    for (int i = 0; i < 8; ++i) r[i] = (bf16)p[i];
#pragma unroll
    for (int i = 0; i < 8; ++i) r[i + 8] = (bf16)p[i + 16];
    return r;
}

__device__ __forceinline__ v8f wmma_bf16(v16bf a, v16bf b, v8f c) {
    return __builtin_amdgcn_wmma_f32_16x16x32_bf16(false, a, false, b, (short)0, c, false, false);
}

// Cooperatively convert ncols x 192 f32 weight rows [col][k] into bf16 LDS panel, stride SXP
__device__ __forceinline__ void stage_weights(bf16* dst, const float* __restrict__ src,
                                              int col0, int ncols, int tid) {
    int total = ncols * 48; // 48 float4 per row
    for (int i = tid; i < total; i += 256) {
        int ln = i / 48, kq = i - ln * 48;
        float4 f = *reinterpret_cast<const float4*>(src + (size_t)(col0 + ln) * C_DIM + kq * 4);
        bf16 pk[4] = {(bf16)f.x, (bf16)f.y, (bf16)f.z, (bf16)f.w};
        *reinterpret_cast<uint2*>(dst + ln * SXP + kq * 4) = *reinterpret_cast<uint2*>(pk);
    }
}

__global__ __launch_bounds__(256)
void wmsa_fused_kernel(const float* __restrict__ x, const float* __restrict__ mask,
                       const float* __restrict__ qkv_w, const float* __restrict__ qkv_b,
                       const float* __restrict__ out_w, const float* __restrict__ out_b,
                       const float* __restrict__ rpb, float* __restrict__ out) {
    __shared__ __align__(16) bf16  sX[NPAD * SXP];        // x tile (bf16), reused as attn-out
    __shared__ __align__(16) bf16  sQ[NPAD * SXP];        // Q (scaled), [row][c]
    __shared__ __align__(16) bf16  sK[NPAD * SXP];        // K, [row][c]
    __shared__ __align__(16) bf16  sVt[C_DIM * SVT];      // V transposed, [c][row]
    __shared__ __align__(16) float sSim[NH * NPAD * SSIM];// logits/probs; aliased as weight panels
    __shared__ float sRpb[169 * NH];                      // relative position bias table

    const int tid  = threadIdx.x;
    const int lane = tid & 31;
    const int wave = tid >> 5;
    const int win  = blockIdx.x;
    const float qk_scale = 0.17677669529663687f; // 32^-0.5

    bf16* sPanel = reinterpret_cast<bf16*>(sSim); // 2 x (64*SXP) bf16 double buffer (51200 B)

    // ---- Phase 1: stage x tile (49x192 f32 -> 64-row bf16, zero-padded), rpb, first weight panel
    const float* xw = x + (size_t)win * (WIN_N * C_DIM);
    for (int i = tid; i < NPAD * 48; i += 256) {
        int r = i / 48, c4 = (i - r * 48) * 4;
        bf16 pk[4];
        if (r < WIN_N) {
            float4 f = *reinterpret_cast<const float4*>(xw + r * C_DIM + c4);
            pk[0] = (bf16)f.x; pk[1] = (bf16)f.y; pk[2] = (bf16)f.z; pk[3] = (bf16)f.w;
        } else {
            pk[0] = pk[1] = pk[2] = pk[3] = (bf16)0.0f;
        }
        *reinterpret_cast<uint2*>(sX + r * SXP + c4) = *reinterpret_cast<uint2*>(pk);
    }
    for (int i = tid; i < 169 * NH; i += 256) sRpb[i] = rpb[i];
    stage_weights(sPanel, qkv_w, 0, 64, tid);
    __syncthreads();

    // ---- Phase 2: QKV = X @ Wqkv^T + b. 9 panels of 64 output cols, software-pipelined.
    // Panel p<3 -> Q cols, 3..5 -> K cols, 6..8 -> V cols (wave-uniform routing).
    {
        const int nl0   = (wave & 3) * 16;       // wave's column tile within panel
        const int mbase = (wave >> 2) * 32;      // wave's two row tiles
        for (int p = 0; p < 9; ++p) {
            bf16* cur = sPanel + (p & 1) * (NPAD * SXP);
            // 1) issue next panel's global loads into registers (latency hidden by WMMAs)
            float4 wreg[12];
            const float* wsrc = qkv_w + (size_t)(p + 1) * 64 * C_DIM;
            if (p + 1 < 9) {
#pragma unroll
                for (int it = 0; it < 12; ++it) {
                    int i = tid + it * 256;
                    int ln = i / 48, kq = i - ln * 48;
                    wreg[it] = *reinterpret_cast<const float4*>(wsrc + ln * C_DIM + kq * 4);
                }
            }
            // 2) preload B fragments (shared by both M tiles of this wave)
            v16bf b[6];
#pragma unroll
            for (int kk = 0; kk < 6; ++kk) b[kk] = load_b_lds(cur, nl0, SXP, kk * 32, lane);
            float bias = qkv_b[p * 64 + nl0 + (lane & 15)];
#pragma unroll
            for (int mt = 0; mt < 2; ++mt) {
                int m0 = mbase + mt * 16;
                v16bf a[6];
#pragma unroll
                for (int kk = 0; kk < 6; ++kk) a[kk] = load_a_lds(sX, m0, SXP, kk * 32, lane);
                v8f acc = {};
#pragma unroll
                for (int kk = 0; kk < 6; ++kk) acc = wmma_bf16(a[kk], b[kk], acc);
                int rb = m0 + ((lane >> 4) << 3);
                if (p < 3) {                              // Q (pre-scaled)
                    int c = p * 64 + nl0 + (lane & 15);
#pragma unroll
                    for (int i = 0; i < 8; ++i)
                        sQ[(rb + i) * SXP + c] = (bf16)((acc[i] + bias) * qk_scale);
                } else if (p < 6) {                       // K
                    int c = (p - 3) * 64 + nl0 + (lane & 15);
#pragma unroll
                    for (int i = 0; i < 8; ++i)
                        sK[(rb + i) * SXP + c] = (bf16)(acc[i] + bias);
                } else {                                  // V, transposed
                    int c = (p - 6) * 64 + nl0 + (lane & 15);
#pragma unroll
                    for (int i = 0; i < 8; ++i)
                        sVt[c * SVT + (rb + i)] = (bf16)(acc[i] + bias);
                }
            }
            // 3) convert + store next panel into the other buffer
            if (p + 1 < 9) {
                bf16* nxt = sPanel + ((p + 1) & 1) * (NPAD * SXP);
#pragma unroll
                for (int it = 0; it < 12; ++it) {
                    int i = tid + it * 256;
                    int ln = i / 48, kq = i - ln * 48;
                    bf16 pk[4] = {(bf16)wreg[it].x, (bf16)wreg[it].y,
                                  (bf16)wreg[it].z, (bf16)wreg[it].w};
                    *reinterpret_cast<uint2*>(nxt + ln * SXP + kq * 4) =
                        *reinterpret_cast<uint2*>(pk);
                }
            }
            __syncthreads();
        }
    }

    // ---- Phase 3: sim = Q_h @ K_h^T. Wave owns (mi, nj-pair); A reused across the pair.
    {
        const int mi  = wave >> 1;          // 0..3
        const int njg = (wave & 1) * 2;     // 0 or 2
#pragma unroll
        for (int h = 0; h < NH; ++h) {
            v16bf a = load_a_lds(sQ, mi * 16, SXP, h * HD, lane);
            float* sp = sSim + h * (NPAD * SSIM);
#pragma unroll
            for (int u = 0; u < 2; ++u) {
                int nj = njg + u;
                v16bf b = load_b_lds(sK, nj * 16, SXP, h * HD, lane);
                v8f acc = {};
                acc = wmma_bf16(a, b, acc);
                int col = nj * 16 + (lane & 15);
                int rb  = mi * 16 + ((lane >> 4) << 3);
#pragma unroll
                for (int i = 0; i < 8; ++i) sp[(rb + i) * SSIM + col] = acc[i];
            }
        }
    }
    __syncthreads();

    // ---- Phase 4: sim += rpb + mask ; softmax over j (logits kept in registers)
    const float* mwin = mask + (size_t)(win & 63) * (WIN_N * WIN_N);
    for (int rr = tid; rr < NH * WIN_N; rr += 256) {
        int h = rr / WIN_N;
        int i = rr - h * WIN_N;
        float* srow = sSim + h * (NPAD * SSIM) + i * SSIM;
        const float* mrow = mwin + i * WIN_N;
        const float* rpbh = sRpb + h;
        int yi = i / 7, xi = i - yi * 7;
        float vals[WIN_N];
        float mx = -3.0e38f;
#pragma unroll
        for (int j = 0; j < WIN_N; ++j) {
            int yj = j / 7, xj = j - yj * 7;
            int ridx = (yi - yj + 6) * 13 + (xi - xj + 6);
            float v = srow[j] + rpbh[ridx * NH] + mrow[j];
            vals[j] = v;
            mx = fmaxf(mx, v);
        }
        float s = 0.0f;
#pragma unroll
        for (int j = 0; j < WIN_N; ++j) {
            float e = __expf(vals[j] - mx);
            vals[j] = e;
            s += e;
        }
        float inv = 1.0f / s;
#pragma unroll
        for (int j = 0; j < WIN_N; ++j) srow[j] = vals[j] * inv;
#pragma unroll
        for (int j = WIN_N; j < NPAD; ++j) srow[j] = 0.0f;
    }
    __syncthreads();

    // ---- Phase 5: out_h = attn_h @ V_h  (wave owns (mi, nj); K=64 -> 2 WMMA per head)
    bf16* sO = sX; // x tile is dead; reuse as attention output (bf16, [row][c], stride SXP)
    {
        const int mi = wave >> 1;   // 0..3
        const int nj = wave & 1;    // 0..1
#pragma unroll
        for (int h = 0; h < NH; ++h) {
            const float* simh = sSim + h * (NPAD * SSIM);
            v8f acc = {};
#pragma unroll
            for (int kk = 0; kk < 2; ++kk) {
                v16bf a = load_a_lds_f32(simh, mi * 16, SSIM, kk * 32, lane);
                v16bf b = load_b_lds(sVt, h * HD + nj * 16, SVT, kk * 32, lane);
                acc = wmma_bf16(a, b, acc);
            }
            int col = h * HD + nj * 16 + (lane & 15);
            int rb  = mi * 16 + ((lane >> 4) << 3);
#pragma unroll
            for (int i = 0; i < 8; ++i) sO[(rb + i) * SXP + col] = (bf16)acc[i];
        }
    }
    __syncthreads();

    // ---- Phase 6: stage out_w (192x192) as bf16 into freed sSim region, then project + bias
    bf16* sW2 = reinterpret_cast<bf16*>(sSim); // 192*SXP bf16 = 76800 B, fits (sim is dead)
    stage_weights(sW2, out_w, 0, C_DIM, tid);
    __syncthreads();

    float* outw = out + (size_t)win * (WIN_N * C_DIM);
    {
        const int m0 = (wave >> 1) * 16;     // wave's row tile
        const int ng = (wave & 1) * 96;      // wave's 6 column tiles
        v16bf a[6];
#pragma unroll
        for (int kk = 0; kk < 6; ++kk) a[kk] = load_a_lds(sO, m0, SXP, kk * 32, lane);
#pragma unroll
        for (int nt = 0; nt < 6; ++nt) {
            int n0 = ng + nt * 16;
            v8f acc = {};
#pragma unroll
            for (int kk = 0; kk < 6; ++kk)
                acc = wmma_bf16(a[kk], load_b_lds(sW2, n0, SXP, kk * 32, lane), acc);
            int col = n0 + (lane & 15);
            int rb  = m0 + ((lane >> 4) << 3);
            float bias = out_b[col];
#pragma unroll
            for (int i = 0; i < 8; ++i) {
                int row = rb + i;
                if (row < WIN_N) outw[row * C_DIM + col] = acc[i] + bias;
            }
        }
    }
}

extern "C" void kernel_launch(void* const* d_in, const int* in_sizes, int n_in,
                              void* d_out, int out_size, void* d_ws, size_t ws_size,
                              hipStream_t stream) {
    (void)n_in; (void)out_size; (void)d_ws; (void)ws_size;
    const float* x     = (const float*)d_in[0];
    const float* mask  = (const float*)d_in[1];
    const float* qkv_w = (const float*)d_in[2];
    const float* qkv_b = (const float*)d_in[3];
    const float* out_w = (const float*)d_in[4];
    const float* out_b = (const float*)d_in[5];
    const float* rpb   = (const float*)d_in[6];
    float* out = (float*)d_out;

    int n_windows = in_sizes[0] / (WIN_N * C_DIM); // 4096
    wmsa_fused_kernel<<<n_windows, 256, 0, stream>>>(x, mask, qkv_w, qkv_b,
                                                     out_w, out_b, rpb, out);
}
